// GATEncoder_88914412961884
// MI455X (gfx1250) — compile-verified, matrix operand order checked by reading, not run on hardware
//
#include <hip/hip_runtime.h>
#include <hip/hip_bf16.h>

// ---------------------------------------------------------------------------
// GAT encoder forward for MI455X (gfx1250, wave32, WMMA).
// GEMMs: operands pre-packed to f16 fragment layout (A row-major Kp-padded,
// B transposed) so the inner loop is 4x global_load_b128 + v_wmma, no branches.
// Attention/softmax on VALU with wave32 shuffle reductions.
// ---------------------------------------------------------------------------

#define HH 4
#define GG 256

typedef __attribute__((ext_vector_type(16))) _Float16 v16h;
typedef __attribute__((ext_vector_type(8)))  _Float16 v8h;
typedef __attribute__((ext_vector_type(8)))  float    v8f;

__device__ __forceinline__ float lrelu(float x) { return x > 0.f ? x : 0.2f * x; }
__device__ __forceinline__ float eluf (float x) { return x > 0.f ? x : (__expf(x) - 1.f); }

__device__ __forceinline__ float wave_max32(float v) {
#pragma unroll
  for (int o = 16; o > 0; o >>= 1) v = fmaxf(v, __shfl_xor(v, o, 32));
  return v;
}
__device__ __forceinline__ float wave_sum32(float v) {
#pragma unroll
  for (int o = 16; o > 0; o >>= 1) v += __shfl_xor(v, o, 32);
  return v;
}

// ---------------------------------------------------------------------------
// Packed-operand GEMM: C[h] = A @ B[h] (+ bias), f32 accumulate.
// Ap: (M x Kp) f16 row-major, Kp % 32 == 0, zero-padded.
// Bp: (Ncp x Kp) f16 = B^T padded, per-head stride strideBhead (halves).
// One wave per 16x16 output tile.
// A fragment (ISA 7.12.2, 16-bit 16x32): lane L holds row m=L&15;
//   elems 0..7  -> K = k0 + (L>>4)*8 + t ; elems 8..15 -> K = k0+16+(L>>4)*8+t
// B fragment: lane L holds column n=L&15, same K pattern.
// C/D v8f: element r -> (row = r + 8*(L>>4), col = L&15).
// ---------------------------------------------------------------------------
__global__ __launch_bounds__(32)
void wmma_gemm_packed(const _Float16* __restrict__ Ap,
                      const _Float16* __restrict__ Bp,
                      float* __restrict__ C,
                      const float* __restrict__ bias,
                      int Kp, int Nc,
                      long strideBhead, long strideChead)
{
  const int lane    = threadIdx.x & 31;
  const int half_id = lane >> 4;
  const int lidx    = lane & 15;
  const int tile_m  = blockIdx.x;
  const int tile_n  = blockIdx.y;
  const int h       = blockIdx.z;

  const _Float16* Arow = Ap + (size_t)(tile_m * 16 + lidx) * Kp;
  const _Float16* Bcol = Bp + (size_t)h * strideBhead + (size_t)(tile_n * 16 + lidx) * Kp;

  v8f acc = {0.f, 0.f, 0.f, 0.f, 0.f, 0.f, 0.f, 0.f};

  for (int k0 = 0; k0 < Kp; k0 += 32) {
    const v8h alo = *(const v8h*)(Arow + k0 + half_id * 8);
    const v8h ahi = *(const v8h*)(Arow + k0 + 16 + half_id * 8);
    const v8h blo = *(const v8h*)(Bcol + k0 + half_id * 8);
    const v8h bhi = *(const v8h*)(Bcol + k0 + 16 + half_id * 8);
    const v16h a = __builtin_shufflevector(alo, ahi, 0,1,2,3,4,5,6,7,8,9,10,11,12,13,14,15);
    const v16h b = __builtin_shufflevector(blo, bhi, 0,1,2,3,4,5,6,7,8,9,10,11,12,13,14,15);
    acc = __builtin_amdgcn_wmma_f32_16x16x32_f16(
        /*neg_a=*/false, a, /*neg_b=*/false, b,
        /*c_mod=*/(short)0, acc, /*reuse_a=*/false, /*reuse_b=*/false);
  }

#pragma unroll
  for (int r = 0; r < 8; ++r) {
    const int orow = tile_m * 16 + r + 8 * half_id;
    const int ocol = tile_n * 16 + lidx;
    if (ocol < Nc) {
      float v = acc[r];
      if (bias) v += bias[ocol];
      C[(size_t)h * strideChead + (size_t)orow * Nc + ocol] = v;
    }
  }
}

// f32 (rows x K) -> f16 (rows x Kp) zero-padded row-major.
__global__ void packA_kernel(const float* __restrict__ A, _Float16* __restrict__ Ap,
                             int rows, int K, int Kp)
{
  const int idx = blockIdx.x * blockDim.x + threadIdx.x;
  if (idx >= rows * Kp) return;
  const int r = idx / Kp, k = idx % Kp;
  Ap[idx] = (k < K) ? (_Float16)A[(size_t)r * K + k] : (_Float16)0.f;
}

// f32 (K x Nc) per head -> f16 B^T (Ncp x Kp) zero-padded.
__global__ void packBT_kernel(const float* __restrict__ B, _Float16* __restrict__ Bp,
                              int K, int Nc, int Kp, int Ncp, long strideBhead)
{
  const int idx = blockIdx.x * blockDim.x + threadIdx.x;
  const int h   = blockIdx.z;
  const int total = Ncp * Kp;
  if (idx >= total) return;
  const int n = idx / Kp, k = idx % Kp;
  const float* Bh = B + (size_t)h * strideBhead;
  Bp[(size_t)h * total + idx] =
      (k < K && n < Nc) ? (_Float16)Bh[(size_t)k * Nc + n] : (_Float16)0.f;
}

// s1[h,i] = Wh[h,i,:] . a[h,0:F];  s2[h,i] = Wh[h,i,:] . a[h,F:2F]
__global__ void dots_kernel(const float* __restrict__ Wh,
                            const float* __restrict__ avec,
                            float* __restrict__ s1, float* __restrict__ s2,
                            int rows, int F, long strideWhHead, long strideAHead)
{
  const int i = blockIdx.x * blockDim.x + threadIdx.x;
  const int h = blockIdx.z;
  if (i >= rows) return;
  const float* W = Wh + (size_t)h * strideWhHead + (size_t)i * F;
  const float* a = avec + (size_t)h * strideAHead;
  float d1 = 0.f, d2 = 0.f;
  for (int f = 0; f < F; ++f) { d1 += W[f] * a[f]; d2 += W[f] * a[F + f]; }
  s1[(size_t)h * rows + i] = d1;
  s2[(size_t)h * rows + i] = d2;
}

__global__ void build_groups_kernel(const int* __restrict__ gid,
                                    int* __restrict__ gcount,
                                    int* __restrict__ members, int N)
{
  const int i = blockIdx.x * blockDim.x + threadIdx.x;
  if (i >= N) return;
  const int g = gid[i];
  if (g > 0 && g <= GG) {
    const int pos = atomicAdd(&gcount[g - 1], 1);
    members[(size_t)(g - 1) * N + pos] = i;
  }
}

// Group-masked attention: wave per (node i, head h). Softmax over group
// members of e_ij = leakyrelu(s1_i + s2_j), then elu(att @ Wh).
// OutT = _Float16 when writing packed GEMM input, float otherwise.
template <typename OutT>
__global__ __launch_bounds__(32)
void attn_group_kernel(const float* __restrict__ Wh,
                       const float* __restrict__ s1,
                       const float* __restrict__ s2,
                       const int* __restrict__ gid,
                       const int* __restrict__ gcount,
                       const int* __restrict__ members,
                       OutT* __restrict__ out,
                       int rows, int F, int outStride)
{
  const int i = blockIdx.x, h = blockIdx.z, lane = threadIdx.x;
  const float* WhH = Wh + (size_t)h * rows * F;
  const float* s2H = s2 + (size_t)h * rows;
  const float  si  = s1[(size_t)h * rows + i];
  const int    g   = gid[i];

  if (g <= 0) {  // self-only: softmax over one element == 1
    for (int f = lane; f < F; f += 32)
      out[(size_t)i * outStride + h * F + f] = (OutT)eluf(WhH[(size_t)i * F + f]);
    return;
  }
  const int  cnt = gcount[g - 1];
  const int* mem = members + (size_t)(g - 1) * rows;

  float m = -3.0e38f;
  for (int idx = lane; idx < cnt; idx += 32)
    m = fmaxf(m, lrelu(si + s2H[mem[idx]]));
  m = wave_max32(m);

  float s = 0.f;
  for (int idx = lane; idx < cnt; idx += 32)
    s += __expf(lrelu(si + s2H[mem[idx]]) - m);
  s = wave_sum32(s);
  const float inv = 1.f / s;

  float accf[3] = {0.f, 0.f, 0.f};   // F <= 96
  for (int idx = 0; idx < cnt; ++idx) {
    const int   j = mem[idx];
    const float w = __expf(lrelu(si + s2H[j]) - m) * inv;
    int r = 0;
    for (int f = lane; f < F; f += 32, ++r)
      accf[r] += w * WhH[(size_t)j * F + f];
  }
  int r = 0;
  for (int f = lane; f < F; f += 32, ++r)
    out[(size_t)i * outStride + h * F + f] = (OutT)eluf(accf[r]);
}

// Dense attention (inter-GAT: adjacency > 0 everywhere). Wave per (i, h).
template <typename OutT>
__global__ __launch_bounds__(32)
void attn_dense_kernel(const float* __restrict__ Wh,
                       const float* __restrict__ s1,
                       const float* __restrict__ s2,
                       OutT* __restrict__ out,
                       int rows, int F, int outStride)
{
  const int i = blockIdx.x, h = blockIdx.z, lane = threadIdx.x;
  const float* WhH = Wh + (size_t)h * rows * F;
  const float* s2H = s2 + (size_t)h * rows;
  const float  si  = s1[(size_t)h * rows + i];

  float m = -3.0e38f;
  for (int j = lane; j < rows; j += 32) m = fmaxf(m, lrelu(si + s2H[j]));
  m = wave_max32(m);
  float s = 0.f;
  for (int j = lane; j < rows; j += 32) s += __expf(lrelu(si + s2H[j]) - m);
  s = wave_sum32(s);
  const float inv = 1.f / s;

  float accf[3] = {0.f, 0.f, 0.f};
  for (int j = 0; j < rows; ++j) {
    const float w = __expf(lrelu(si + s2H[j]) - m) * inv;
    int r = 0;
    for (int f = lane; f < F; f += 32, ++r)
      accf[r] += w * WhH[(size_t)j * F + f];
  }
  int r = 0;
  for (int f = lane; f < F; f += 32, ++r)
    out[(size_t)i * outStride + h * F + f] = (OutT)eluf(accf[r]);
}

// Row-wise log_softmax, F <= 32. Wave per row.
__global__ __launch_bounds__(32)
void log_softmax_kernel(const float* __restrict__ in, float* __restrict__ out, int F)
{
  const int i = blockIdx.x, lane = threadIdx.x;
  const float v = (lane < F) ? in[(size_t)i * F + lane] : -3.0e38f;
  const float m = wave_max32(v);
  const float e = (lane < F) ? __expf(v - m) : 0.f;
  const float s = wave_sum32(e);
  if (lane < F) out[(size_t)i * F + lane] = v - m - __logf(s);
}

// g_in[g] = mean over members of x (R @ x), written f16 padded to Kp cols.
__global__ __launch_bounds__(32)
void group_mean_pack_kernel(const float* __restrict__ x,
                            const int* __restrict__ gcount,
                            const int* __restrict__ members,
                            _Float16* __restrict__ out, int N, int F, int Kp)
{
  const int g = blockIdx.x, lane = threadIdx.x;
  const int cnt = gcount[g];
  const int* mem = members + (size_t)g * N;
  float acc = 0.f;
  if (lane < F) {
    for (int idx = 0; idx < cnt; ++idx) acc += x[(size_t)mem[idx] * F + lane];
  }
  const float v = (lane < F && cnt > 0) ? acc / (float)cnt : 0.f;
  if (lane < Kp) out[(size_t)g * Kp + lane] = (_Float16)v;
}

// xcat = [x_intra | R.T @ group_out] written f16 packed (N x 2F).
__global__ void xcat_pack_kernel(const float* __restrict__ x_intra,
                                 const float* __restrict__ group_out,
                                 const int* __restrict__ gid,
                                 const int* __restrict__ gcount,
                                 _Float16* __restrict__ xcatp, int N, int F)
{
  const int i = blockIdx.x * blockDim.x + threadIdx.x;
  if (i >= N) return;
  const int g = gid[i];
  float invc = 0.f;
  const float* go = group_out;
  if (g > 0) {
    const int c = gcount[g - 1];
    invc = (c > 0) ? 1.f / (float)c : 0.f;
    go = group_out + (size_t)(g - 1) * F;
  }
  for (int f = 0; f < F; ++f) {
    xcatp[(size_t)i * (2 * F) + f]     = (_Float16)x_intra[(size_t)i * F + f];
    xcatp[(size_t)i * (2 * F) + F + f] = (_Float16)((g > 0) ? go[f] * invc : 0.f);
  }
}

extern "C" void kernel_launch(void* const* d_in, const int* in_sizes, int n_in,
                              void* d_out, int out_size, void* d_ws, size_t ws_size,
                              hipStream_t stream)
{
  const float* h_states = (const float*)d_in[0];   // (N, 40)
  const float* W_intra  = (const float*)d_in[2];   // (H, 40, 72)
  const float* a_intra  = (const float*)d_in[3];   // (H, 144, 1)
  const float* Wo_intra = (const float*)d_in[4];   // (288, 16)
  const float* ao_intra = (const float*)d_in[5];   // (32, 1)
  const float* W_inter  = (const float*)d_in[6];   // (H, 16, 72)
  const float* a_inter  = (const float*)d_in[7];   // (H, 144, 1)
  const float* Wo_inter = (const float*)d_in[8];   // (288, 16)
  const float* ao_inter = (const float*)d_in[9];   // (32, 1)
  const float* W_emb    = (const float*)d_in[10];  // (32, 24)
  const float* b_emb    = (const float*)d_in[11];  // (24,)
  const int*   end_grp  = (const int*)d_in[13];    // (N, 1)

  const int N = in_sizes[0] / 40;                  // 4096

  // ---- workspace carve-up ----
  float* ws = (float*)d_ws;
  size_t o = 0;
  // f32 buffers
  float* Wh_i    = ws + o; o += (size_t)HH * N * 72;
  float* Wh2     = ws + o; o += (size_t)N * 16;
  float* out1    = ws + o; o += (size_t)N * 16;
  float* x_intra = ws + o; o += (size_t)N * 16;
  float* Whg     = ws + o; o += (size_t)HH * GG * 72;
  float* Wh2g    = ws + o; o += (size_t)GG * 16;
  float* out1g   = ws + o; o += (size_t)GG * 16;
  float* g_out   = ws + o; o += (size_t)GG * 16;
  float* s1_i    = ws + o; o += (size_t)HH * N;
  float* s2_i    = ws + o; o += (size_t)HH * N;
  float* s1_2    = ws + o; o += (size_t)N;
  float* s2_2    = ws + o; o += (size_t)N;
  float* s1g     = ws + o; o += (size_t)HH * GG;
  float* s2g     = ws + o; o += (size_t)HH * GG;
  float* s1g2    = ws + o; o += (size_t)GG;
  float* s2g2    = ws + o; o += (size_t)GG;
  // f16 packed buffers (offsets rounded to 8 floats = 32B for b128 loads)
  auto allocH = [&](size_t nHalves) -> _Float16* {
    o = (o + 7) & ~(size_t)7;
    _Float16* p = (_Float16*)(ws + o);
    o += (nHalves + 1) / 2;
    return p;
  };
  _Float16* Ap_x       = allocH((size_t)N * 64);        // h_states, Kp=64
  _Float16* x2p        = allocH((size_t)N * 288);       // head outputs, Kp=288
  _Float16* g_inp      = allocH((size_t)GG * 32);       // group means, Kp=32
  _Float16* x2gp       = allocH((size_t)GG * 288);      // inter head outputs
  _Float16* xcatp      = allocH((size_t)N * 32);        // concat, Kp=32
  _Float16* Bp_Wintra  = allocH((size_t)HH * 80 * 64);  // (Ncp=80, Kp=64) x H
  _Float16* Bp_Wointra = allocH((size_t)16 * 288);
  _Float16* Bp_Winter  = allocH((size_t)HH * 80 * 32);
  _Float16* Bp_Wointer = allocH((size_t)16 * 288);
  _Float16* Bp_Wemb    = allocH((size_t)32 * 32);
  // int buffers
  o = (o + 7) & ~(size_t)7;
  int* gcount  = (int*)(ws + o); o += GG;
  int* members = (int*)(ws + o); o += (size_t)GG * N;

  // ---- group structure ----
  hipMemsetAsync(gcount, 0, GG * sizeof(int), stream);
  build_groups_kernel<<<(N + 255) / 256, 256, 0, stream>>>(end_grp, gcount, members, N);

  // ---- operand packing (weights + input features) ----
  packA_kernel<<<(N * 64 + 255) / 256, 256, 0, stream>>>(h_states, Ap_x, N, 40, 64);
  packBT_kernel<<<dim3((80 * 64 + 255) / 256, 1, HH), 256, 0, stream>>>(
      W_intra, Bp_Wintra, 40, 72, 64, 80, (long)40 * 72);
  packBT_kernel<<<dim3((16 * 288 + 255) / 256, 1, 1), 256, 0, stream>>>(
      Wo_intra, Bp_Wointra, 288, 16, 288, 16, 0);
  packBT_kernel<<<dim3((80 * 32 + 255) / 256, 1, HH), 256, 0, stream>>>(
      W_inter, Bp_Winter, 16, 72, 32, 80, (long)16 * 72);
  packBT_kernel<<<dim3((16 * 288 + 255) / 256, 1, 1), 256, 0, stream>>>(
      Wo_inter, Bp_Wointer, 288, 16, 288, 16, 0);
  packBT_kernel<<<dim3((32 * 32 + 255) / 256, 1, 1), 256, 0, stream>>>(
      W_emb, Bp_Wemb, 32, 24, 32, 32, 0);

  // ---- intra GAT, head layer: Wh = X @ W_intra[h]  (4096x40 @ 40x72) ----
  wmma_gemm_packed<<<dim3(N / 16, 5, HH), 32, 0, stream>>>(
      Ap_x, Bp_Wintra, Wh_i, nullptr, 64, 72, (long)80 * 64, (long)N * 72);
  dots_kernel<<<dim3((N + 127) / 128, 1, HH), 128, 0, stream>>>(
      Wh_i, a_intra, s1_i, s2_i, N, 72, (long)N * 72, 144);
  attn_group_kernel<_Float16><<<dim3(N, 1, HH), 32, 0, stream>>>(
      Wh_i, s1_i, s2_i, end_grp, gcount, members, x2p, N, 72, 288);

  // ---- intra GAT, output layer: Wh2 = x2 @ Wo_intra  (4096x288 @ 288x16) ----
  wmma_gemm_packed<<<dim3(N / 16, 1, 1), 32, 0, stream>>>(
      x2p, Bp_Wointra, Wh2, nullptr, 288, 16, 0, 0);
  dots_kernel<<<dim3((N + 127) / 128, 1, 1), 128, 0, stream>>>(
      Wh2, ao_intra, s1_2, s2_2, N, 16, 0, 0);
  attn_group_kernel<float><<<dim3(N, 1, 1), 32, 0, stream>>>(
      Wh2, s1_2, s2_2, end_grp, gcount, members, out1, N, 16, 16);
  log_softmax_kernel<<<N, 32, 0, stream>>>(out1, x_intra, 16);

  // ---- group pooling: g_in = R @ x_intra  (256x16, packed f16 Kp=32) ----
  group_mean_pack_kernel<<<GG, 32, 0, stream>>>(x_intra, gcount, members, g_inp, N, 16, 32);

  // ---- inter GAT, head layer: Whg = g_in @ W_inter[h]  (256x16 @ 16x72) ----
  wmma_gemm_packed<<<dim3(GG / 16, 5, HH), 32, 0, stream>>>(
      g_inp, Bp_Winter, Whg, nullptr, 32, 72, (long)80 * 32, (long)GG * 72);
  dots_kernel<<<dim3((GG + 127) / 128, 1, HH), 128, 0, stream>>>(
      Whg, a_inter, s1g, s2g, GG, 72, (long)GG * 72, 144);
  attn_dense_kernel<_Float16><<<dim3(GG, 1, HH), 32, 0, stream>>>(
      Whg, s1g, s2g, x2gp, GG, 72, 288);

  // ---- inter GAT, output layer ----
  wmma_gemm_packed<<<dim3(GG / 16, 1, 1), 32, 0, stream>>>(
      x2gp, Bp_Wointer, Wh2g, nullptr, 288, 16, 0, 0);
  dots_kernel<<<dim3((GG + 127) / 128, 1, 1), 128, 0, stream>>>(
      Wh2g, ao_inter, s1g2, s2g2, GG, 16, 0, 0);
  attn_dense_kernel<float><<<dim3(GG, 1, 1), 32, 0, stream>>>(
      Wh2g, s1g2, s2g2, out1g, GG, 16, 16);
  log_softmax_kernel<<<GG, 32, 0, stream>>>(out1g, g_out, 16);

  // ---- scatter + concat + final projection into d_out (N x 24) ----
  xcat_pack_kernel<<<(N + 255) / 256, 256, 0, stream>>>(
      x_intra, g_out, end_grp, gcount, xcatp, N, 16);
  wmma_gemm_packed<<<dim3(N / 16, 2, 1), 32, 0, stream>>>(
      xcatp, Bp_Wemb, (float*)d_out, b_emb, 32, 24, 0, 0);
}
